// MultiHeadAttention_26439818674765
// MI455X (gfx1250) — compile-verified, matrix operand order checked by reading, not run on hardware
//
#include <hip/hip_runtime.h>
#include <hip/hip_bf16.h>

typedef _Float16 half_t;
typedef __attribute__((ext_vector_type(16))) _Float16 v16h;
typedef __attribute__((ext_vector_type(8)))  _Float16 v8h;
typedef __attribute__((ext_vector_type(8)))  float    v8f;

#define DIMX 1024
#define HEADS 16
#define DIM_HEAD 64
#define SEQ 2048
#define BATCH 2
#define INNER (HEADS * DIM_HEAD)   // 1024
#define NROWS (BATCH * SEQ)        // 4096
#define QKSCALE 0.125f             // 1/sqrt(64)

// CDNA5 async global->LDS copies (signature verified by compile probe:
// pointers to GCC-style `int __vector(4)` in AS1 / AS3).
#if defined(__HIP_DEVICE_COMPILE__) &&                                        \
    __has_builtin(__builtin_amdgcn_global_load_async_to_lds_b128) &&          \
    __has_builtin(__builtin_amdgcn_s_wait_asynccnt)
#define HAVE_ASYNC_LDS 1
#else
#define HAVE_ASYNC_LDS 0
#endif

#if HAVE_ASYNC_LDS
typedef int async_i4 __attribute__((vector_size(16)));
__device__ __forceinline__ void async_copy_b128(const half_t* g, half_t* l) {
  __builtin_amdgcn_global_load_async_to_lds_b128(
      (__attribute__((address_space(1))) async_i4*)g,
      (__attribute__((address_space(3))) async_i4*)l, 0, 0);
}
__device__ __forceinline__ void wait_async() {
  __builtin_amdgcn_s_wait_asynccnt(0);
}
#else
__device__ __forceinline__ void wait_async() {}
#endif

// ---------------------------------------------------------------------------
// Vectorized WMMA fragment load (CDNA5 wave32, 16-bit operands).
// Lane L -> row = L&15, kb = (L>>4)*8; 16 halves are two contiguous 16B runs:
// [row*ld+kb .. +7] and [row*ld+kb+16 .. +23].  Serves A operands (row-major
// MxK) and B operands stored K-contiguous per output column (NxK).
// ---------------------------------------------------------------------------
__device__ __forceinline__ v16h load_frag_rm(const half_t* __restrict__ s,
                                             int ld) {
  const int lane = threadIdx.x & 31;
  const half_t* p = s + (lane & 15) * ld + ((lane >> 4) << 3);
  const v8h lo = *(const v8h*)p;
  const v8h hi = *(const v8h*)(p + 16);
  return __builtin_shufflevector(lo, hi, 0, 1, 2, 3, 4, 5, 6, 7, 8, 9, 10, 11,
                                 12, 13, 14, 15);
}

__device__ __forceinline__ v8f wmma16(v16h a, v16h b, v8f c) {
  return __builtin_amdgcn_wmma_f32_16x16x32_f16(false, a, false, b, (short)0, c,
                                                false, false);
}

// ---------------------------------------------------------------------------
// Kernel: f32 [K][N] -> f16 [N][K] transpose-cast (weights, one time).
// ---------------------------------------------------------------------------
__global__ __launch_bounds__(256)
void cast_transpose_kernel(const float* __restrict__ in, half_t* __restrict__ out,
                           int K, int N) {
  __shared__ half_t tile[32][33];
  const int k0 = blockIdx.y * 32, n0 = blockIdx.x * 32;
  const int tx = threadIdx.x & 31, ty = threadIdx.x >> 5;
#pragma unroll
  for (int r = ty; r < 32; r += 8)
    tile[r][tx] = (half_t)in[(size_t)(k0 + r) * N + n0 + tx];
  __syncthreads();
#pragma unroll
  for (int r = ty; r < 32; r += 8)
    out[(size_t)(n0 + r) * K + k0 + tx] = tile[tx][r];
}

// ---------------------------------------------------------------------------
// Kernel: LayerNorm row (DIMX=1024) + cast to f16.  One 256-thread block/row.
// ---------------------------------------------------------------------------
__global__ __launch_bounds__(256)
void ln_cast_kernel(const float* __restrict__ x, const float* __restrict__ gamma,
                    const float* __restrict__ beta, half_t* __restrict__ xn) {
  const int row = blockIdx.x;
  const float* xr = x + (size_t)row * DIMX;
  float v[4], s = 0.f, ss = 0.f;
#pragma unroll
  for (int i = 0; i < 4; ++i) {
    v[i] = xr[threadIdx.x + i * 256];
    s += v[i];
    ss += v[i] * v[i];
  }
#pragma unroll
  for (int m = 16; m >= 1; m >>= 1) {
    s += __shfl_xor(s, m, 32);
    ss += __shfl_xor(ss, m, 32);
  }
  __shared__ float red[18];
  const int wave = threadIdx.x >> 5;
  if ((threadIdx.x & 31) == 0) {
    red[wave] = s;
    red[wave + 8] = ss;
  }
  __syncthreads();
  if (threadIdx.x == 0) {
    float ts = 0.f, tss = 0.f;
#pragma unroll
    for (int w = 0; w < 8; ++w) { ts += red[w]; tss += red[w + 8]; }
    const float mu = ts * (1.f / DIMX);
    red[16] = mu;
    red[17] = rsqrtf(tss * (1.f / DIMX) - mu * mu + 1e-5f);
  }
  __syncthreads();
  const float mu = red[16], rs = red[17];
  half_t* xo = xn + (size_t)row * DIMX;
#pragma unroll
  for (int i = 0; i < 4; ++i) {
    const int c = threadIdx.x + i * 256;
    xo[c] = (half_t)(((v[i] - mu) * rs) * gamma[c] + beta[c]);
  }
}

// ---------------------------------------------------------------------------
// WMMA GEMM skeleton: 256 thr (8 waves, 2x4), tile 64(M) x 256(N), BK=32,
// double-buffered LDS with async global->LDS staging (software pipelined).
// ---------------------------------------------------------------------------
#define BM 64
#define BN 256
#define BK 32
#define KDIM 1024   // K-dim is 1024 for both GEMMs

__global__ __launch_bounds__(256)
void gemm_qkv_kernel(const half_t* __restrict__ A, const half_t* __restrict__ Bt,
                     half_t* __restrict__ Qb, half_t* __restrict__ Kb,
                     half_t* __restrict__ Vb) {
  __shared__ half_t As[2][BM * BK];
  __shared__ half_t Bs[2][BN * BK];
  const int tid = threadIdx.x;
  const int lane = tid & 31, wave = tid >> 5;
  const int wM = wave >> 2, wN = wave & 3;
  const int m0 = blockIdx.y * BM;
  const int n0 = blockIdx.x * BN;

  v8f acc[2][4];
#pragma unroll
  for (int i = 0; i < 2; ++i)
#pragma unroll
    for (int j = 0; j < 4; ++j) acc[i][j] = {};

  const int ar = tid >> 2, ac = (tid & 3) << 3;   // A: 8 halves/thread
  const half_t* agp = A + (size_t)(m0 + ar) * KDIM + ac;
  const half_t* bgp = Bt + (size_t)(n0 + tid) * KDIM;  // row n = tid

  auto stage = [&](int k0, int buf) {
#if HAVE_ASYNC_LDS
    async_copy_b128(agp + k0, &As[buf][ar * BK + ac]);
#pragma unroll
    for (int u = 0; u < 4; ++u)
      async_copy_b128(bgp + k0 + u * 8, &Bs[buf][tid * BK + u * 8]);
#else
    const uint4 ta = *(const uint4*)(agp + k0);
    uint4 tb[4];
#pragma unroll
    for (int u = 0; u < 4; ++u) tb[u] = *(const uint4*)(bgp + k0 + u * 8);
    *(uint4*)&As[buf][ar * BK + ac] = ta;
#pragma unroll
    for (int u = 0; u < 4; ++u) *(uint4*)&Bs[buf][tid * BK + u * 8] = tb[u];
#endif
  };

  stage(0, 0);
  wait_async();
  __syncthreads();

  for (int k0 = 0; k0 < KDIM; k0 += BK) {
    const int buf = (k0 >> 5) & 1;
    if (k0 + BK < KDIM) stage(k0 + BK, buf ^ 1);
    v16h af[2], bf[4];
#pragma unroll
    for (int i = 0; i < 2; ++i)
      af[i] = load_frag_rm(&As[buf][(wM * 32 + i * 16) * BK], BK);
#pragma unroll
    for (int j = 0; j < 4; ++j)
      bf[j] = load_frag_rm(&Bs[buf][(wN * 64 + j * 16) * BK], BK);
#pragma unroll
    for (int i = 0; i < 2; ++i)
#pragma unroll
      for (int j = 0; j < 4; ++j) acc[i][j] = wmma16(af[i], bf[j], acc[i][j]);
    if (k0 + BK < KDIM) wait_async();
    __syncthreads();
  }

  // Epilogue: which/h/d uniform per 16-col fragment (q/k/v splits at 1024
  // boundaries are 16-aligned); b/sidx uniform per 8-row half-fragment.
  const int nlane = lane & 15, hi = lane >> 4;
#pragma unroll
  for (int j = 0; j < 4; ++j) {
    const int gn = n0 + wN * 64 + j * 16 + nlane;
    const int which = gn >> 10;      // 0:q 1:k 2:v
    const int rem = gn & 1023;
    const int h = rem >> 6, d = rem & 63;
#pragma unroll
    for (int i = 0; i < 2; ++i) {
      const int gm0 = m0 + wM * 32 + i * 16 + hi * 8;  // 8-aligned row base
      const int b = gm0 >> 11, s0 = gm0 & 2047;
      const size_t bh = (size_t)b * HEADS + h;
      if (which == 0) {
        half_t* p = Qb + (bh * SEQ + s0) * DIM_HEAD + d;
#pragma unroll
        for (int r = 0; r < 8; ++r)
          p[(size_t)r * DIM_HEAD] = (half_t)(acc[i][j][r] * QKSCALE);
      } else if (which == 1) {
        half_t* p = Kb + (bh * SEQ + s0) * DIM_HEAD + d;
#pragma unroll
        for (int r = 0; r < 8; ++r)
          p[(size_t)r * DIM_HEAD] = (half_t)acc[i][j][r];
      } else {  // V stored transposed [bh][d][s]
        half_t* p = Vb + (bh * DIM_HEAD + d) * SEQ + s0;
#pragma unroll
        for (int r = 0; r < 8; ++r) p[r] = (half_t)acc[i][j][r];
      }
    }
  }
}

__global__ __launch_bounds__(256)
void gemm_out_kernel(const half_t* __restrict__ A, const half_t* __restrict__ Bt,
                     const float* __restrict__ bias, float* __restrict__ out) {
  __shared__ half_t As[2][BM * BK];
  __shared__ half_t Bs[2][BN * BK];
  const int tid = threadIdx.x;
  const int lane = tid & 31, wave = tid >> 5;
  const int wM = wave >> 2, wN = wave & 3;
  const int m0 = blockIdx.y * BM;
  const int n0 = blockIdx.x * BN;

  v8f acc[2][4];
#pragma unroll
  for (int i = 0; i < 2; ++i)
#pragma unroll
    for (int j = 0; j < 4; ++j) acc[i][j] = {};

  const int ar = tid >> 2, ac = (tid & 3) << 3;
  const half_t* agp = A + (size_t)(m0 + ar) * KDIM + ac;
  const half_t* bgp = Bt + (size_t)(n0 + tid) * KDIM;

  auto stage = [&](int k0, int buf) {
#if HAVE_ASYNC_LDS
    async_copy_b128(agp + k0, &As[buf][ar * BK + ac]);
#pragma unroll
    for (int u = 0; u < 4; ++u)
      async_copy_b128(bgp + k0 + u * 8, &Bs[buf][tid * BK + u * 8]);
#else
    const uint4 ta = *(const uint4*)(agp + k0);
    uint4 tb[4];
#pragma unroll
    for (int u = 0; u < 4; ++u) tb[u] = *(const uint4*)(bgp + k0 + u * 8);
    *(uint4*)&As[buf][ar * BK + ac] = ta;
#pragma unroll
    for (int u = 0; u < 4; ++u) *(uint4*)&Bs[buf][tid * BK + u * 8] = tb[u];
#endif
  };

  stage(0, 0);
  wait_async();
  __syncthreads();

  for (int k0 = 0; k0 < KDIM; k0 += BK) {
    const int buf = (k0 >> 5) & 1;
    if (k0 + BK < KDIM) stage(k0 + BK, buf ^ 1);
    v16h af[2], bf[4];
#pragma unroll
    for (int i = 0; i < 2; ++i)
      af[i] = load_frag_rm(&As[buf][(wM * 32 + i * 16) * BK], BK);
#pragma unroll
    for (int j = 0; j < 4; ++j)
      bf[j] = load_frag_rm(&Bs[buf][(wN * 64 + j * 16) * BK], BK);
#pragma unroll
    for (int i = 0; i < 2; ++i)
#pragma unroll
      for (int j = 0; j < 4; ++j) acc[i][j] = wmma16(af[i], bf[j], acc[i][j]);
    if (k0 + BK < KDIM) wait_async();
    __syncthreads();
  }

  const int nlane = lane & 15, hi = lane >> 4;
#pragma unroll
  for (int j = 0; j < 4; ++j) {
    const int gn = n0 + wN * 64 + j * 16 + nlane;
    const float bv = bias[gn];
#pragma unroll
    for (int i = 0; i < 2; ++i) {
      const int gm0 = m0 + wM * 32 + i * 16 + hi * 8;
      float* p = out + (size_t)gm0 * DIMX + gn;
#pragma unroll
      for (int r = 0; r < 8; ++r) p[(size_t)r * DIMX] = acc[i][j][r] + bv;
    }
  }
}

// ---------------------------------------------------------------------------
// Flash attention (causal, online softmax).  One block = one (b,h, 64-query
// tile); 4 waves, 16 query rows each.  K [j][d] and V [d][j] tiles staged in
// double-buffered LDS with async copies, pipelined across key tiles.
// ---------------------------------------------------------------------------
__global__ __launch_bounds__(128)
void flash_attn_kernel(const half_t* __restrict__ Qb, const half_t* __restrict__ Kb,
                       const half_t* __restrict__ Vt, half_t* __restrict__ Ob) {
  __shared__ half_t Ks[2][64 * 64];   // [j][d]
  __shared__ half_t Vs[2][64 * 64];   // [d][j]
  __shared__ half_t Ps[4][16 * 64];

  const int tid = threadIdx.x, lane = tid & 31, wave = tid >> 5;
  const int qt = blockIdx.x & 31;        // SEQ/64 tiles
  const int bh = blockIdx.x >> 5;        // b*HEADS + h
  const size_t base = (size_t)bh * SEQ * DIM_HEAD;
  const int rowbase = qt * 64 + wave * 16;
  const int nlane = lane & 15, hi = lane >> 4;

  // Q fragments for this wave's 16 rows (global b128 loads, once)
  v16h aq[2];
  {
    const half_t* qs = Qb + base + (size_t)rowbase * DIM_HEAD;
    aq[0] = load_frag_rm(qs, DIM_HEAD);
    aq[1] = load_frag_rm(qs + 32, DIM_HEAD);
  }

  v8f o[4];
#pragma unroll
  for (int nf = 0; nf < 4; ++nf) o[nf] = {};
  float mi[8], li[8];
#pragma unroll
  for (int r = 0; r < 8; ++r) { mi[r] = -3e38f; li[r] = 0.f; }

  const int tr = tid >> 1, tc = (tid & 1) << 5;  // coop tile load: 32 halves each
  const half_t* kgp = Kb + base + (size_t)tr * DIM_HEAD + tc;
  const half_t* vgp = Vt + base + (size_t)tr * SEQ + tc;

  auto stage_kv = [&](int jt, int buf) {
    const half_t* kg = kgp + (size_t)jt * 64 * DIM_HEAD;
    const half_t* vg = vgp + jt * 64;
    half_t* kl = &Ks[buf][tr * 64 + tc];
    half_t* vl = &Vs[buf][tr * 64 + tc];
#if HAVE_ASYNC_LDS
#pragma unroll
    for (int u = 0; u < 4; ++u) {
      async_copy_b128(kg + u * 8, kl + u * 8);
      async_copy_b128(vg + u * 8, vl + u * 8);
    }
#else
    uint4 tk[4], tv[4];
#pragma unroll
    for (int u = 0; u < 4; ++u) {
      tk[u] = *(const uint4*)(kg + u * 8);
      tv[u] = *(const uint4*)(vg + u * 8);
    }
#pragma unroll
    for (int u = 0; u < 4; ++u) {
      *(uint4*)(kl + u * 8) = tk[u];
      *(uint4*)(vl + u * 8) = tv[u];
    }
#endif
  };

  stage_kv(0, 0);
  wait_async();
  __syncthreads();

  for (int jt = 0; jt <= qt; ++jt) {
    const int buf = jt & 1;
    if (jt < qt) stage_kv(jt + 1, buf ^ 1);

    // S = Q @ K^T   (16 x 64 per wave); B operand from Ks[j][d], reduce over d
    v8f s[4];
#pragma unroll
    for (int nf = 0; nf < 4; ++nf) {
      v8f z = {};
      z = wmma16(aq[0], load_frag_rm(&Ks[buf][(nf * 16) * 64], 64), z);
      z = wmma16(aq[1], load_frag_rm(&Ks[buf][(nf * 16) * 64 + 32], 64), z);
      s[nf] = z;
    }

    // causal mask (only tiles touching the diagonal)
    if (jt * 64 + 63 > rowbase) {
#pragma unroll
      for (int nf = 0; nf < 4; ++nf) {
        const int col = jt * 64 + nf * 16 + nlane;
#pragma unroll
        for (int r = 0; r < 8; ++r)
          if (col > rowbase + r + hi * 8) s[nf][r] = -3e38f;
      }
    }

    // online softmax per row (16 lanes of a half-wave share a row)
    float alpha[8];
#pragma unroll
    for (int r = 0; r < 8; ++r) {
      float mx = s[0][r];
#pragma unroll
      for (int nf = 1; nf < 4; ++nf) mx = fmaxf(mx, s[nf][r]);
#pragma unroll
      for (int m = 8; m >= 1; m >>= 1) mx = fmaxf(mx, __shfl_xor(mx, m, 32));
      const float mnew = fmaxf(mi[r], mx);
      alpha[r] = __expf(mi[r] - mnew);
      float rs = 0.f;
#pragma unroll
      for (int nf = 0; nf < 4; ++nf) {
        const float p = __expf(s[nf][r] - mnew);
        s[nf][r] = p;
        rs += p;
      }
#pragma unroll
      for (int m = 8; m >= 1; m >>= 1) rs += __shfl_xor(rs, m, 32);
      li[r] = li[r] * alpha[r] + rs;
      mi[r] = mnew;
    }
#pragma unroll
    for (int nf = 0; nf < 4; ++nf)
#pragma unroll
      for (int r = 0; r < 8; ++r) o[nf][r] *= alpha[r];

    // P (C layout) -> per-wave LDS region -> A-fragment layout.
    // Same-wave LDS ops complete in order; a compiler barrier is enough.
    half_t* P = &Ps[wave][0];
#pragma unroll
    for (int nf = 0; nf < 4; ++nf)
#pragma unroll
      for (int r = 0; r < 8; ++r)
        P[(r + hi * 8) * 64 + nf * 16 + nlane] = (half_t)s[nf][r];
    __asm__ volatile("" ::: "memory");

    const v16h ap0 = load_frag_rm(P, 64);
    const v16h ap1 = load_frag_rm(P + 32, 64);
#pragma unroll
    for (int nf = 0; nf < 4; ++nf) {
      // B operand from Vs[d][j]: n = d (nf*16..), k = j chunks 0..31 / 32..63
      o[nf] = wmma16(ap0, load_frag_rm(&Vs[buf][(nf * 16) * 64], 64), o[nf]);
      o[nf] = wmma16(ap1, load_frag_rm(&Vs[buf][(nf * 16) * 64 + 32], 64), o[nf]);
    }

    if (jt < qt) wait_async();
    __syncthreads();
  }

  // normalize + store [b][s][h*64+d] as f16 for the output GEMM
  const int b = bh >> 4, h = bh & 15;
  float invl[8];
#pragma unroll
  for (int r = 0; r < 8; ++r) invl[r] = 1.0f / li[r];
#pragma unroll
  for (int nf = 0; nf < 4; ++nf) {
    const int col = h * 64 + nf * 16 + nlane;
    half_t* p = Ob + ((size_t)b * SEQ + rowbase + hi * 8) * INNER + col;
#pragma unroll
    for (int r = 0; r < 8; ++r)
      p[(size_t)r * INNER] = (half_t)(o[nf][r] * invl[r]);
  }
}

// ---------------------------------------------------------------------------
extern "C" void kernel_launch(void* const* d_in, const int* in_sizes, int n_in,
                              void* d_out, int out_size, void* d_ws, size_t ws_size,
                              hipStream_t stream) {
  const float* x    = (const float*)d_in[0];
  const float* lng  = (const float*)d_in[1];
  const float* lnb  = (const float*)d_in[2];
  const float* wqkv = (const float*)d_in[3];
  const float* wout = (const float*)d_in[4];
  const float* bout = (const float*)d_in[5];
  float* out = (float*)d_out;

  char* ws = (char*)d_ws;
  half_t* xn    = (half_t*)ws;  ws += (size_t)NROWS * DIMX * sizeof(half_t);
  half_t* qb    = (half_t*)ws;  ws += (size_t)NROWS * INNER * sizeof(half_t);
  half_t* kb    = (half_t*)ws;  ws += (size_t)NROWS * INNER * sizeof(half_t);
  half_t* vt    = (half_t*)ws;  ws += (size_t)NROWS * INNER * sizeof(half_t);
  half_t* attn  = (half_t*)ws;  ws += (size_t)NROWS * INNER * sizeof(half_t);
  half_t* wq_t  = (half_t*)ws;  ws += (size_t)DIMX * 3 * INNER * sizeof(half_t);
  half_t* wo_t  = (half_t*)ws;  ws += (size_t)INNER * DIMX * sizeof(half_t);

  // 1) transpose-cast weights to f16 [N][K]
  cast_transpose_kernel<<<dim3(3 * INNER / 32, DIMX / 32), 256, 0, stream>>>(
      wqkv, wq_t, DIMX, 3 * INNER);
  cast_transpose_kernel<<<dim3(DIMX / 32, INNER / 32), 256, 0, stream>>>(
      wout, wo_t, INNER, DIMX);
  // 2) LayerNorm + cast
  ln_cast_kernel<<<NROWS, 256, 0, stream>>>(x, lng, lnb, xn);
  // 3) QKV projection (WMMA), head-major scatter, scale folded into Q, V transposed
  gemm_qkv_kernel<<<dim3(3 * INNER / BN, NROWS / BM), 256, 0, stream>>>(
      xn, wq_t, qb, kb, vt);
  // 4) causal flash attention (WMMA)
  flash_attn_kernel<<<BATCH * HEADS * (SEQ / 64), 128, 0, stream>>>(qb, kb, vt,
                                                                    attn);
  // 5) output projection + bias (WMMA)
  gemm_out_kernel<<<dim3(DIMX / BN, NROWS / BM), 256, 0, stream>>>(attn, wo_t,
                                                                   bout, out);
}
